// NNConv_15101105013036
// MI455X (gfx1250) — compile-verified, matrix operand order checked by reading, not run on hardware
//
#include <hip/hip_runtime.h>
#include <hip/hip_bf16.h>

typedef __attribute__((ext_vector_type(16))) _Float16 v16h;
typedef __attribute__((ext_vector_type(8)))  _Float16 v8h;
typedef __attribute__((ext_vector_type(2)))  _Float16 h2;
typedef __attribute__((ext_vector_type(8)))  float    v8f;

#define IN_C  16
#define OUT_C 16
#define EDGE_DIM 8
#define HID   32
// GEMM per 16-edge tile: [16 x 512] (P) @ [512 x 16] (W2 repacked) -> [16 x 16]
// K = HID*IN_C = 512 -> 16 chunks of 32, plus 1 extra chunk for the b2 term.
#define NCHUNK 16
#define BCHUNK_HALFS 512            // 32 lanes * 16 f16 per B fragment chunk
#define LDS_B_HALFS ((NCHUNK + 1) * BCHUNK_HALFS)

union HalfPair { unsigned int u; _Float16 f[2]; h2 v; };
union AFrag    { v16h v; h2 p[8]; };
union XPack    { v8h v; h2 p[4]; };

// ---------------------------------------------------------------------------
// Kernel 1: out[n, o] = bias[o] + sum_i x[n,i] * root[i*16+o]   (fp32 exact)
// ---------------------------------------------------------------------------
__global__ void nnconv_root_kernel(const float* __restrict__ x,
                                   const float* __restrict__ root,
                                   const float* __restrict__ bias,
                                   float* __restrict__ out, int N) {
    int t = blockIdx.x * blockDim.x + threadIdx.x;
    if (t >= N * OUT_C) return;
    int n = t >> 4;
    int o = t & 15;
    float a = bias[o];
    const float* xr = x + (size_t)n * IN_C;
#pragma unroll
    for (int i = 0; i < IN_C; ++i) a = fmaf(xr[i], root[i * OUT_C + o], a);
    out[t] = a;
}

// ---------------------------------------------------------------------------
// Kernel 2: per-16-edge tile WMMA GEMM + atomic scatter-add
// ---------------------------------------------------------------------------
__global__ void __launch_bounds__(256)
nnconv_edges_kernel(const float* __restrict__ x,
                    const int*   __restrict__ ei,      // [2, E]: row then col
                    const float* __restrict__ pseudo,  // [E, 8]
                    const float* __restrict__ W1,      // [8, 32]
                    const float* __restrict__ b1,      // [32]
                    const float* __restrict__ W2,      // [32, 256]
                    const float* __restrict__ b2,      // [256]
                    float* __restrict__ out,           // [N, 16]
                    int E, int numTiles, int numWaves) {
    __shared__ alignas(32) _Float16 sB[LDS_B_HALFS];   // prepacked B fragments
    __shared__ float sW1[EDGE_DIM * HID];
    __shared__ float sb1[HID];

    const int tid = threadIdx.x;

    // ---- prepack W2 into the WMMA B-fragment layout (f16), once per block ----
    // B chunk kc, lane L (n = L&15, hi = L>>4), slot s (0..15):
    //   j = s + 8*(hi + (s>=8)), k = 2*kc + (s>=8), i = j & 15
    //   value = W2[k*256 + i*16 + n]
    for (int idx = tid; idx < NCHUNK * BCHUNK_HALFS; idx += 256) {
        int kc = idx >> 9;
        int L  = (idx >> 4) & 31;
        int s  = idx & 15;
        int hi = L >> 4;
        int n  = L & 15;
        int j  = s + 8 * (hi + ((s >= 8) ? 1 : 0));
        int k  = 2 * kc + ((s >= 8) ? 1 : 0);
        int i  = j & 15;
        sB[idx] = (_Float16)W2[k * 256 + i * 16 + n];
    }
    // extra chunk (index NCHUNK): b2 term. B(j,o) = j<16 ? b2[j*16+o] : 0
    for (int idx = tid; idx < BCHUNK_HALFS; idx += 256) {
        int L  = (idx >> 4) & 31;
        int s  = idx & 15;
        int hi = L >> 4;
        int n  = L & 15;
        int j  = s + 8 * (hi + ((s >= 8) ? 1 : 0));
        sB[NCHUNK * BCHUNK_HALFS + idx] =
            (j < IN_C) ? (_Float16)b2[j * 16 + n] : (_Float16)0.0f;
    }
    if (tid < EDGE_DIM * HID) sW1[tid] = W1[tid];
    if (tid < HID)            sb1[tid] = b1[tid];
    __syncthreads();

    const int lane = tid & 31;
    const int wave = tid >> 5;
    const int waveGlobal = blockIdx.x * (blockDim.x >> 5) + wave;
    const int m  = lane & 15;   // edge-in-tile for A fragment
    const int hi = lane >> 4;   // which K half-group this lane carries
    const int o  = lane & 15;   // output channel for D fragment

    // bpermute byte-indices: owner of k<16 pairs is lane m, k>=16 is lane m+16
    const int selLo = m * 4;
    const int selHi = (m + 16) * 4;

    for (int t = waveGlobal; t < numTiles; t += numWaves) {
        const int eBase = t * 16;
        const int e     = eBase + m;
        const bool valid = (e < E);
        const int esafe  = valid ? e : 0;

        // x[col[e], hi*8 .. hi*8+7] as packed f16 pairs
        const int colIdx = ei[E + esafe];
        const float* xp = x + (size_t)colIdx * IN_C + hi * 8;
        XPack xpk;
#pragma unroll
        for (int i = 0; i < 8; ++i)
            xpk.v[i] = valid ? (_Float16)xp[i] : (_Float16)0.0f;

        // pseudo row
        float ps[EDGE_DIM];
        const float* pp = pseudo + (size_t)esafe * EDGE_DIM;
#pragma unroll
        for (int i = 0; i < EDGE_DIM; ++i) ps[i] = valid ? pp[i] : 0.0f;

        // h[m, hi*16 .. hi*16+15]: each pair-lane computes HALF the hidden
        // units (128 FMAs); partner halves gathered below via ds_bpermute.
        float ha[16];
#pragma unroll
        for (int k = 0; k < 16; ++k) ha[k] = sb1[hi * 16 + k];
#pragma unroll
        for (int d = 0; d < EDGE_DIM; ++d) {
            float pd = ps[d];
            const float4* wrow = (const float4*)&sW1[d * HID + hi * 16];
#pragma unroll
            for (int q = 0; q < 4; ++q) {
                float4 wv = wrow[q];
                ha[4 * q + 0] = fmaf(pd, wv.x, ha[4 * q + 0]);
                ha[4 * q + 1] = fmaf(pd, wv.y, ha[4 * q + 1]);
                ha[4 * q + 2] = fmaf(pd, wv.z, ha[4 * q + 2]);
                ha[4 * q + 3] = fmaf(pd, wv.w, ha[4 * q + 3]);
            }
        }
        // relu + pack to f16 pairs (one dword = h[2c], h[2c+1] of this half)
        unsigned int ownPk[8];
#pragma unroll
        for (int c = 0; c < 8; ++c) {
            HalfPair cv;
            cv.f[0] = (_Float16)fmaxf(ha[2 * c + 0], 0.0f);
            cv.f[1] = (_Float16)fmaxf(ha[2 * c + 1], 0.0f);
            ownPk[c] = cv.u;
        }
        // hAll[c] = packed (h[2c], h[2c+1]) in GLOBAL k order, gathered
        // straight from the owning lane (no selects needed).
        unsigned int hAll[16];
#pragma unroll
        for (int c = 0; c < 8; ++c) {
            hAll[c]     = (unsigned int)__builtin_amdgcn_ds_bpermute(selLo, (int)ownPk[c]);
            hAll[c + 8] = (unsigned int)__builtin_amdgcn_ds_bpermute(selHi, (int)ownPk[c]);
        }

        // ---- main GEMM: msg_tile = P @ W2p, K = 512 (+16 for b2) ----
        v8f acc = {};
#pragma unroll
        for (int kc = 0; kc < NCHUNK; ++kc) {
            HalfPair cv; cv.u = hAll[kc];
            h2 hl = { cv.f[0], cv.f[0] };   // splat h[2kc]
            h2 hh = { cv.f[1], cv.f[1] };   // splat h[2kc+1]
            AFrag af;
#pragma unroll
            for (int tt = 0; tt < 4; ++tt) {
                af.p[tt]     = hl * xpk.p[tt];   // v_pk_mul_f16
                af.p[tt + 4] = hh * xpk.p[tt];   // v_pk_mul_f16
            }
            v16h b = *(const v16h*)&sB[kc * BCHUNK_HALFS + lane * 16];
            acc = __builtin_amdgcn_wmma_f32_16x16x32_f16(
                false, af.v, false, b, (short)0, acc, false, false);
        }
        {   // b2 chunk: A = [x_col | 0]
            AFrag af;
            h2 z = { (_Float16)0.0f, (_Float16)0.0f };
#pragma unroll
            for (int tt = 0; tt < 4; ++tt) {
                af.p[tt]     = xpk.p[tt];
                af.p[tt + 4] = z;
            }
            v16h b = *(const v16h*)&sB[NCHUNK * BCHUNK_HALFS + lane * 16];
            acc = __builtin_amdgcn_wmma_f32_16x16x32_f16(
                false, af.v, false, b, (short)0, acc, false, false);
        }

        // ---- scatter-add: D fragment lane=o, VGPR r -> edge (hi*8 + r) ----
#pragma unroll
        for (int r = 0; r < 8; ++r) {
            int m2 = hi * 8 + r;
            int e2 = eBase + m2;
            if (e2 < E) {
                int rowIdx = ei[e2];
                atomicAdd(&out[(size_t)rowIdx * OUT_C + o], acc[r]);
            }
        }
    }
}

extern "C" void kernel_launch(void* const* d_in, const int* in_sizes, int n_in,
                              void* d_out, int out_size, void* d_ws, size_t ws_size,
                              hipStream_t stream) {
    (void)n_in; (void)out_size; (void)d_ws; (void)ws_size;
    const float* x      = (const float*)d_in[0];
    const int*   ei     = (const int*)  d_in[1];
    const float* pseudo = (const float*)d_in[2];
    const float* W1     = (const float*)d_in[3];
    const float* b1     = (const float*)d_in[4];
    const float* W2     = (const float*)d_in[5];
    const float* b2     = (const float*)d_in[6];
    const float* root   = (const float*)d_in[7];
    const float* bias   = (const float*)d_in[8];
    float* out = (float*)d_out;

    const int N = in_sizes[0] / IN_C;      // 32000
    const int E = in_sizes[1] / 2;         // 512000
    const int numTiles = (E + 15) / 16;

    // Kernel 1: seed out with root transform + bias (fp32).
    int rootThreads = N * OUT_C;
    nnconv_root_kernel<<<dim3((rootThreads + 255) / 256), dim3(256), 0, stream>>>(
        x, root, bias, out, N);

    // Kernel 2: fused edge MLP + bilinear WMMA GEMM + atomic scatter.
    const int wavesPerBlock = 8;           // 256 threads = 8 wave32
    int blocks = (numTiles + wavesPerBlock - 1) / wavesPerBlock;
    if (blocks > 2048) blocks = 2048;
    const int numWaves = blocks * wavesPerBlock;
    nnconv_edges_kernel<<<dim3(blocks), dim3(256), 0, stream>>>(
        x, ei, pseudo, W1, b1, W2, b2, out, E, numTiles, numWaves);
}
